// MaskDecoderMamba_58944131170406
// MI455X (gfx1250) — matrix operand order repaired and checked by reading, not compiled, not run on hardware
//
#include <hip/hip_runtime.h>
#include <cstdint>
#include <cstddef>

// ---------------------------------------------------------------------------
// MaskDecoderMamba for MI455X (gfx1250, wave32, WMMA).
// All GEMM-shaped work (in_proj / x_proj / out_proj / 3x3x3 conv as implicit
// GEMM) runs through v_wmma_f32_16x16x32_bf16. The 3^3 conv (~0.93 TFLOP)
// dominates; conv kernel uses LDS-staged activations (ds_load path) +
// 2x4 register blocking so the WMMA pipe, not L1/L2, is the limiter.
// ---------------------------------------------------------------------------

typedef __attribute__((ext_vector_type(16))) __bf16 v16bf;
typedef __attribute__((ext_vector_type(8)))  __bf16 v8bf;
typedef __attribute__((ext_vector_type(8)))  float  v8f;

#define L_SEQ   4096
#define DMODEL  256
#define DINNER  512
#define DSTATE  16
#define DTRANK  16
#define V64     262144    // 64^3
#define V128    2097152   // 128^3
#define NSTATB  4096      // instance-norm partial blocks (V64/64)

__device__ __forceinline__ float siluf(float x) { return x / (1.f + __expf(-x)); }

// ---------------------------------------------------------------------------
// Generic bf16 WMMA GEMM:  C[M x N] = A[M x K] * W[N x K]^T   (all row-major)
// Fragment layouts used here (contiguous k order, same for A and B):
//   A 16x32 bf16 : lane<16 holds row m=lane, K k0..k0+15
//                  lane>=16 holds row m=lane-16, K k0+16..k0+31
//   B 32x16 bf16 : lane holds col n=lane&15, K half = (lane<16 ? 0..15 : 16..31)
//   C/D 16x16 f32: lane l, vgpr v -> row m = v + 8*(l>>4), col n = l&15
// ---------------------------------------------------------------------------
__global__ void k_gemm_bf16(const __bf16* __restrict__ A, const __bf16* __restrict__ W,
                            float* __restrict__ C, int M, int N, int K,
                            int lda, int ldw, int ldc)
{
    const int lane = threadIdx.x & 31;
    const int mrow = lane & 15;
    const int hiK  = lane >> 4;
    const int nt = N >> 4;
    const int total = (M >> 4) * nt;
    int wt = blockIdx.x * (blockDim.x >> 5) + (threadIdx.x >> 5);
    const int wstride = gridDim.x * (blockDim.x >> 5);
    for (; wt < total; wt += wstride) {
        const int ntile = wt % nt;
        const int mtile = wt / nt;
        const __bf16* __restrict__ Arow = A + (size_t)(mtile * 16 + mrow) * lda;
        const __bf16* __restrict__ Wrow = W + (size_t)(ntile * 16 + mrow) * ldw;
        v8f acc = {0.f, 0.f, 0.f, 0.f, 0.f, 0.f, 0.f, 0.f};
        for (int k0 = 0; k0 < K; k0 += 32) {
            const v8bf alo = *(const v8bf*)(Arow + k0 + hiK * 16);
            const v8bf ahi = *(const v8bf*)(Arow + k0 + hiK * 16 + 8);
            v16bf a;
#pragma unroll
            for (int i = 0; i < 8; ++i) { a[i] = alo[i]; a[i + 8] = ahi[i]; }
            const v16bf b = *(const v16bf*)(Wrow + k0 + hiK * 16);
            acc = __builtin_amdgcn_wmma_f32_16x16x32_bf16(
                false, a, false, b, (short)0, acc, false, false);
        }
#pragma unroll
        for (int v = 0; v < 8; ++v)
            C[(size_t)(mtile * 16 + 8 * hiK + v) * ldc + ntile * 16 + mrow] = acc[v];
    }
}

// ---- small prep kernels ----------------------------------------------------

// seq_bf[l][c] = bf16(x[c][l])   (x is [256][4096] f32)
__global__ void k_seq_cast(const float* __restrict__ x, __bf16* __restrict__ seq_bf)
{
    int i = blockIdx.x * blockDim.x + threadIdx.x;
    if (i >= L_SEQ * DMODEL) return;
    int c = i & (DMODEL - 1);
    int l = i >> 8;
    seq_bf[i] = (__bf16)x[(size_t)c * L_SEQ + l];
}

__global__ void k_cast_bf16(const float* __restrict__ src, __bf16* __restrict__ dst, int n)
{
    int i = blockIdx.x * blockDim.x + threadIdx.x;
    if (i < n) dst[i] = (__bf16)src[i];
}

// causal depthwise conv1d (k=4, pad left 3) + bias + SiLU; xi = xz[:, :512]
__global__ void k_conv1d_silu(const float* __restrict__ xz,
                              const float* __restrict__ w, const float* __restrict__ b,
                              float* __restrict__ xc, __bf16* __restrict__ xc_bf)
{
    int idx = blockIdx.x * blockDim.x + threadIdx.x;
    if (idx >= L_SEQ * DINNER) return;
    int d = idx & (DINNER - 1);
    int l = idx >> 9;
    float acc = b[d];
#pragma unroll
    for (int t = 0; t < 4; ++t) {
        int l2 = l + t - 3;
        if (l2 >= 0) acc += xz[(size_t)l2 * 1024 + d] * w[d * 4 + t];
    }
    float s = siluf(acc);
    xc[idx] = s;
    xc_bf[idx] = (__bf16)s;
}

// delta = softplus(dt @ dt_proj_w^T + b), dt = dbl[:, :16]
__global__ void k_delta(const float* __restrict__ dbl, const float* __restrict__ dtw,
                        const float* __restrict__ dtb, float* __restrict__ delta)
{
    __shared__ float sdt[DTRANK];
    const int l = blockIdx.x;
    const int d = threadIdx.x;
    if (d < DTRANK) sdt[d] = dbl[(size_t)l * 48 + d];
    __syncthreads();
    float acc = dtb[d];
#pragma unroll
    for (int r = 0; r < DTRANK; ++r) acc += sdt[r] * dtw[d * DTRANK + r];
    delta[(size_t)l * DINNER + d] = (acc > 20.f) ? acc : log1pf(__expf(acc));
}

// selective scan: block = 32 lanes = 2 channels x 16 states; sequential over L.
__global__ void k_scan(const float* __restrict__ delta, const float* __restrict__ xc,
                       const float* __restrict__ dbl, const float* __restrict__ A_log,
                       float* __restrict__ yscan)
{
    const int lane = threadIdx.x & 31;
    const int s = lane & 15;
    const int d = blockIdx.x * 2 + (lane >> 4);
    const float a = -__expf(A_log[d * DSTATE + s]);
    float h = 0.f;
    for (int l = 0; l < L_SEQ; ++l) {
        const float dl = delta[(size_t)l * DINNER + d];
        const float xv = xc[(size_t)l * DINNER + d];
        const float Bv = dbl[(size_t)l * 48 + 16 + s];
        const float Cv = dbl[(size_t)l * 48 + 32 + s];
        h = __expf(dl * a) * h + dl * Bv * xv;
        float p = h * Cv;
#pragma unroll
        for (int off = 8; off > 0; off >>= 1) p += __shfl_xor(p, off, 16);
        if (s == 0) yscan[(size_t)l * DINNER + d] = p;
    }
}

// y = (yscan + D*xc) * silu(z); z = xz[:, 512:]
__global__ void k_gate(const float* __restrict__ yscan, const float* __restrict__ xc,
                       const float* __restrict__ xz, const float* __restrict__ Dskip,
                       __bf16* __restrict__ y_bf)
{
    int idx = blockIdx.x * blockDim.x + threadIdx.x;
    if (idx >= L_SEQ * DINNER) return;
    int d = idx & (DINNER - 1);
    int l = idx >> 9;
    float zv = xz[(size_t)l * 1024 + 512 + d];
    float y = (yscan[idx] + Dskip[d] * xc[idx]) * siluf(zv);
    y_bf[idx] = (__bf16)y;
}

// trilinear 16^3 -> 64^3, align_corners; mout is [16^3 vox][256] (channels-last).
// One block per output voxel, 256 threads over channels. Writes bf16 CL volume.
__global__ void k_up16to64(const float* __restrict__ mout, __bf16* __restrict__ volCL)
{
    const int b = blockIdx.x;
    const int c = threadIdx.x;
    const int ox = b & 63, oy = (b >> 6) & 63, oz = b >> 12;
    const float sc = 15.f / 63.f;
    const float fz = oz * sc, fy = oy * sc, fx = ox * sc;
    const int z0 = (int)fz, y0 = (int)fy, x0 = (int)fx;
    const float wz = fz - z0, wy = fy - y0, wx = fx - x0;
    const int z1 = z0 + 1 > 15 ? 15 : z0 + 1;
    const int y1 = y0 + 1 > 15 ? 15 : y0 + 1;
    const int x1 = x0 + 1 > 15 ? 15 : x0 + 1;
    auto at = [&](int z, int y, int x) {
        return mout[(size_t)(((z * 16 + y) * 16 + x)) * 256 + c];
    };
    float c00 = at(z0, y0, x0) * (1.f - wx) + at(z0, y0, x1) * wx;
    float c01 = at(z0, y1, x0) * (1.f - wx) + at(z0, y1, x1) * wx;
    float c10 = at(z1, y0, x0) * (1.f - wx) + at(z1, y0, x1) * wx;
    float c11 = at(z1, y1, x0) * (1.f - wx) + at(z1, y1, x1) * wx;
    float v = (c00 * (1.f - wy) + c01 * wy) * (1.f - wz) +
              (c10 * (1.f - wy) + c11 * wy) * wz;
    volCL[(size_t)b * 256 + c] = (__bf16)v;
}

// pack cls_w1 [cout][cin][3][3][3] f32 -> wpack[off][cout][cin] bf16
__global__ void k_pack_w1(const float* __restrict__ w1, __bf16* __restrict__ wpack)
{
    int i = blockIdx.x * blockDim.x + threadIdx.x;
    if (i >= 27 * 256 * 256) return;
    int off = i >> 16;
    int rem = i & 65535;
    int cout = rem >> 8;
    int cin = rem & 255;
    wpack[(size_t)off * 65536 + rem] = (__bf16)w1[((size_t)cout * 256 + cin) * 27 + off];
}

// ---------------------------------------------------------------------------
// 3x3x3 conv 256->256 as implicit GEMM through WMMA, LDS-staged activations.
// Block (256 thr / 8 waves) owns one full x-row (64 voxels) at fixed (y,z),
// all 256 couts. For each of 9 (dz,dy) slabs, the haloed row (66 vox x 256
// cin, bf16, exact zero halo) is staged in LDS once; waves consume it for
// 3 dx taps x 8 K-chunks. Wave tile = 2 M-tiles x 4 N-tiles (8 acc tiles):
// each A fragment feeds 4 WMMAs, each B fragment feeds 2.
// ---------------------------------------------------------------------------
__global__ void k_conv3d_wmma(const __bf16* __restrict__ volCL,
                              const __bf16* __restrict__ wpack,
                              float* __restrict__ h)
{
    __shared__ __attribute__((aligned(16))) __bf16 sA[66 * 256];   // 33 KB

    const int tid  = threadIdx.x;
    const int lane = tid & 31;
    const int wv   = tid >> 5;        // 0..7
    const int mg   = wv & 1;          // which 32-voxel half of the row
    const int ng   = wv >> 1;         // which 64-cout quarter
    const int mrow = lane & 15;
    const int hiK  = lane >> 4;
    const int y = blockIdx.x & 63;
    const int z = blockIdx.x >> 6;

    v8f acc[2][4];
#pragma unroll
    for (int mt = 0; mt < 2; ++mt)
#pragma unroll
        for (int nt = 0; nt < 4; ++nt)
#pragma unroll
            for (int v = 0; v < 8; ++v) acc[mt][nt][v] = 0.f;

    for (int dzy = 0; dzy < 9; ++dzy) {
        const int dz = dzy / 3 - 1;
        const int dy = dzy % 3 - 1;
        const int zz = z + dz, yy = y + dy;
        if (((unsigned)zz >= 64u) | ((unsigned)yy >= 64u)) continue;  // zero slab

        __syncthreads();   // previous slab fully consumed
        {   // cooperative stage: 64 voxels * 256 cin -> sA[256..16639]; halo = 0
            const uint4* __restrict__ src =
                (const uint4*)(volCL + (size_t)((zz * 64 + yy) * 64) * 256);
            uint4* dst = (uint4*)(sA + 256);
            for (int i = tid; i < 2048; i += 256) dst[i] = src[i];
            const uint4 z4 = {0u, 0u, 0u, 0u};
            if (tid < 32) ((uint4*)sA)[tid] = z4;                       // x = -1
            else if (tid < 64) ((uint4*)(sA + 256 + 16384))[tid - 32] = z4; // x = 64
        }
        __syncthreads();

#pragma unroll
        for (int dx = 0; dx < 3; ++dx) {
            const int off = (dz + 1) * 9 + (dy + 1) * 3 + dx;
            const __bf16* __restrict__ wbase =
                wpack + (size_t)off * 65536 + (size_t)(ng * 64 + mrow) * 256 + hiK * 16;
            __builtin_prefetch((const void*)wbase, 0, 1);   // global_prefetch_b8
            for (int k0 = 0; k0 < 256; k0 += 32) {
                v16bf a[2];
#pragma unroll
                for (int mt = 0; mt < 2; ++mt) {
                    const int xidx = mg * 32 + mt * 16 + mrow + dx;  // 0..65
                    const __bf16* ap = sA + xidx * 256 + k0 + hiK * 16;
                    const v8bf lo = *(const v8bf*)ap;        // ds_load_b128
                    const v8bf hi = *(const v8bf*)(ap + 8);
#pragma unroll
                    for (int i = 0; i < 8; ++i) { a[mt][i] = lo[i]; a[mt][i + 8] = hi[i]; }
                }
#pragma unroll
                for (int nt = 0; nt < 4; ++nt) {
                    const v16bf b = *(const v16bf*)(wbase + (size_t)nt * 16 * 256 + k0);
                    acc[0][nt] = __builtin_amdgcn_wmma_f32_16x16x32_bf16(
                        false, a[0], false, b, (short)0, acc[0][nt], false, false);
                    acc[1][nt] = __builtin_amdgcn_wmma_f32_16x16x32_bf16(
                        false, a[1], false, b, (short)0, acc[1][nt], false, false);
                }
            }
        }
    }

    const int voxbase = (z * 64 + y) * 64 + mg * 32;
#pragma unroll
    for (int mt = 0; mt < 2; ++mt)
#pragma unroll
        for (int nt = 0; nt < 4; ++nt)
#pragma unroll
            for (int v = 0; v < 8; ++v)
                h[(size_t)(voxbase + mt * 16 + 8 * hiK + v) * 256 +
                  ng * 64 + nt * 16 + mrow] = acc[mt][nt][v];
}

// instance-norm pass 1: per-block (64 voxels) per-channel partial sums (coalesced).
__global__ void k_norm_partials(const float* __restrict__ h,
                                float* __restrict__ psum, float* __restrict__ psq)
{
    const int b = blockIdx.x;     // 4096 blocks
    const int c = threadIdx.x;    // 256 channels
    float s = 0.f, q = 0.f;
    const float* __restrict__ hp = h + (size_t)b * 64 * 256 + c;
    for (int v = 0; v < 64; ++v) {
        float x = hp[(size_t)v * 256];
        s += x;
        q += x * x;
    }
    psum[(size_t)b * 256 + c] = s;
    psq[(size_t)b * 256 + c] = q;
}

// pass 2: fixed-order tree reduction (deterministic) -> mu, rsigma per channel
__global__ void k_norm_reduce(const float* __restrict__ psum, const float* __restrict__ psq,
                              float* __restrict__ mu, float* __restrict__ rs)
{
    const int c = blockIdx.x;
    const int t = threadIdx.x;
    __shared__ float ss[256], sq[256];
    float a = 0.f, b = 0.f;
    for (int i = t; i < NSTATB; i += 256) {
        a += psum[(size_t)i * 256 + c];
        b += psq[(size_t)i * 256 + c];
    }
    ss[t] = a; sq[t] = b;
    __syncthreads();
    for (int o = 128; o > 0; o >>= 1) {
        if (t < o) { ss[t] += ss[t + o]; sq[t] += sq[t + o]; }
        __syncthreads();
    }
    if (t == 0) {
        float m = ss[0] / (float)V64;
        float v = sq[0] / (float)V64 - m * m;
        mu[c] = m;
        rs[c] = rsqrtf(v + 1e-5f);
    }
}

// norm + relu + 1x1 conv (256 -> 2) fused; out2 is [2][64^3] f32
__global__ void k_cls(const float* __restrict__ h, const float* __restrict__ mu,
                      const float* __restrict__ rs, const float* __restrict__ w2,
                      const float* __restrict__ b2, float* __restrict__ out2)
{
    __shared__ float sm[256], sr[256], wa[256], wb[256];
    const int t = threadIdx.x;
    sm[t] = mu[t]; sr[t] = rs[t]; wa[t] = w2[t]; wb[t] = w2[256 + t];
    __syncthreads();
    const int vox = blockIdx.x * 256 + t;
    const float* __restrict__ hp = h + (size_t)vox * 256;
    float a0 = 0.f, a1 = 0.f;
    for (int c = 0; c < 256; ++c) {
        float v = (hp[c] - sm[c]) * sr[c];
        v = fmaxf(v, 0.f);
        a0 += v * wa[c];
        a1 += v * wb[c];
    }
    out2[vox] = a0 + b2[0];
    out2[V64 + vox] = a1 + b2[1];
}

// final trilinear 64^3 -> 128^3, align_corners; out [2][128][128][128] f32
__global__ void k_up_final(const float* __restrict__ out2, float* __restrict__ out)
{
    int idx = blockIdx.x * blockDim.x + threadIdx.x;
    if (idx >= 2 * V128) return;
    const int cls = idx >> 21;
    const int rem = idx & (V128 - 1);
    const int ox = rem & 127, oy = (rem >> 7) & 127, oz = rem >> 14;
    const float sc = 63.f / 127.f;
    const float fz = oz * sc, fy = oy * sc, fx = ox * sc;
    const int z0 = (int)fz, y0 = (int)fy, x0 = (int)fx;
    const float wz = fz - z0, wy = fy - y0, wx = fx - x0;
    const int z1 = z0 + 1 > 63 ? 63 : z0 + 1;
    const int y1 = y0 + 1 > 63 ? 63 : y0 + 1;
    const int x1 = x0 + 1 > 63 ? 63 : x0 + 1;
    const float* __restrict__ p = out2 + (size_t)cls * V64;
    auto at = [&](int z, int y, int x) { return p[(z * 64 + y) * 64 + x]; };
    float c00 = at(z0, y0, x0) * (1.f - wx) + at(z0, y0, x1) * wx;
    float c01 = at(z0, y1, x0) * (1.f - wx) + at(z0, y1, x1) * wx;
    float c10 = at(z1, y0, x0) * (1.f - wx) + at(z1, y0, x1) * wx;
    float c11 = at(z1, y1, x0) * (1.f - wx) + at(z1, y1, x1) * wx;
    out[idx] = (c00 * (1.f - wy) + c01 * wy) * (1.f - wz) +
               (c10 * (1.f - wy) + c11 * wy) * wz;
}

// ---------------------------------------------------------------------------

extern "C" void kernel_launch(void* const* d_in, const int* in_sizes, int n_in,
                              void* d_out, int out_size, void* d_ws, size_t ws_size,
                              hipStream_t stream)
{
    const float* x     = (const float*)d_in[0];
    const float* in_w  = (const float*)d_in[1];   // [1024][256]
    const float* c1w   = (const float*)d_in[2];   // [512][1][4]
    const float* c1b   = (const float*)d_in[3];   // [512]
    const float* xp_w  = (const float*)d_in[4];   // [48][512]
    const float* dt_w  = (const float*)d_in[5];   // [512][16]
    const float* dt_b  = (const float*)d_in[6];   // [512]
    const float* A_log = (const float*)d_in[7];   // [512][16]
    const float* Dskip = (const float*)d_in[8];   // [512]
    const float* out_w = (const float*)d_in[9];   // [256][512]
    const float* w1    = (const float*)d_in[10];  // [256][256][3][3][3]
    const float* w2    = (const float*)d_in[11];  // [2][256]
    const float* b2    = (const float*)d_in[12];  // [2]
    float* out = (float*)d_out;

    // linear workspace carve (~475 MB total; h f32 dominates)
    char* p = (char*)d_ws;
    auto carve = [&](size_t bytes) -> char* {
        char* r = p;
        p += (bytes + 255) & ~(size_t)255;
        return r;
    };
    __bf16* seq_bf   = (__bf16*)carve((size_t)L_SEQ * DMODEL * 2);
    __bf16* w_in_bf  = (__bf16*)carve((size_t)1024 * 256 * 2);
    __bf16* w_xp_bf  = (__bf16*)carve((size_t)48 * 512 * 2);
    __bf16* w_out_bf = (__bf16*)carve((size_t)256 * 512 * 2);
    float*  xz       = (float*)carve((size_t)L_SEQ * 1024 * 4);
    float*  xc       = (float*)carve((size_t)L_SEQ * DINNER * 4);
    __bf16* xc_bf    = (__bf16*)carve((size_t)L_SEQ * DINNER * 2);
    float*  dbl      = (float*)carve((size_t)L_SEQ * 48 * 4);
    float*  delta    = (float*)carve((size_t)L_SEQ * DINNER * 4);
    float*  yscan    = (float*)carve((size_t)L_SEQ * DINNER * 4);
    __bf16* y_bf     = (__bf16*)carve((size_t)L_SEQ * DINNER * 2);
    float*  mout     = (float*)carve((size_t)L_SEQ * DMODEL * 4);
    __bf16* volCL    = (__bf16*)carve((size_t)V64 * 256 * 2);
    __bf16* wpack    = (__bf16*)carve((size_t)27 * 65536 * 2);
    float*  h        = (float*)carve((size_t)V64 * 256 * 4);
    float*  psum     = (float*)carve((size_t)NSTATB * 256 * 4);
    float*  psq      = (float*)carve((size_t)NSTATB * 256 * 4);
    float*  mu       = (float*)carve(256 * 4);
    float*  rs       = (float*)carve(256 * 4);
    float*  out2     = (float*)carve((size_t)2 * V64 * 4);
    (void)ws_size; (void)in_sizes; (void)n_in; (void)out_size;

    // ---- Mamba block ----
    k_seq_cast<<<(L_SEQ * DMODEL + 255) / 256, 256, 0, stream>>>(x, seq_bf);
    k_cast_bf16<<<(1024 * 256 + 255) / 256, 256, 0, stream>>>(in_w, w_in_bf, 1024 * 256);
    k_cast_bf16<<<(48 * 512 + 255) / 256, 256, 0, stream>>>(xp_w, w_xp_bf, 48 * 512);
    k_cast_bf16<<<(256 * 512 + 255) / 256, 256, 0, stream>>>(out_w, w_out_bf, 256 * 512);

    // in_proj: [4096,256] x [1024,256]^T -> xz [4096,1024]; 16384 wave tiles
    k_gemm_bf16<<<2048, 256, 0, stream>>>(seq_bf, w_in_bf, xz,
                                          L_SEQ, 1024, 256, 256, 256, 1024);
    k_conv1d_silu<<<(L_SEQ * DINNER + 255) / 256, 256, 0, stream>>>(xz, c1w, c1b, xc, xc_bf);
    // x_proj: [4096,512] x [48,512]^T -> dbl [4096,48]; 768 wave tiles
    k_gemm_bf16<<<96, 256, 0, stream>>>(xc_bf, w_xp_bf, dbl,
                                        L_SEQ, 48, 512, 512, 512, 48);
    k_delta<<<L_SEQ, DINNER, 0, stream>>>(dbl, dt_w, dt_b, delta);
    k_scan<<<DINNER / 2, 32, 0, stream>>>(delta, xc, dbl, A_log, yscan);
    k_gate<<<(L_SEQ * DINNER + 255) / 256, 256, 0, stream>>>(yscan, xc, xz, Dskip, y_bf);
    // out_proj: [4096,512] x [256,512]^T -> mout [4096,256]; 4096 wave tiles
    k_gemm_bf16<<<512, 256, 0, stream>>>(y_bf, w_out_bf, mout,
                                         L_SEQ, 256, 512, 512, 512, 256);

    // ---- decoder head ----
    k_up16to64<<<V64, 256, 0, stream>>>(mout, volCL);
    k_pack_w1<<<(27 * 256 * 256 + 255) / 256, 256, 0, stream>>>(w1, wpack);
    // conv3d: one block per (z,y) x-row; 8 waves = 2 M-groups x 4 N-groups
    k_conv3d_wmma<<<4096, 256, 0, stream>>>(volCL, wpack, h);
    k_norm_partials<<<NSTATB, 256, 0, stream>>>(h, psum, psq);
    k_norm_reduce<<<256, 256, 0, stream>>>(psum, psq, mu, rs);
    k_cls<<<V64 / 256, 256, 0, stream>>>(h, mu, rs, w2, b2, out2);
    k_up_final<<<(2 * V128 + 255) / 256, 256, 0, stream>>>(out2, out);
}
